// TFST_62938450755992
// MI455X (gfx1250) — matrix operand order repaired and checked by reading, not compiled, not run on hardware
//
#include <hip/hip_runtime.h>
#include <math.h>

// ---------------------------------------------------------------------------
// CQT-magnitude as a WMMA GEMM on gfx1250 (wave32).
//   out[s][f][t] = | sum_k kernel_s[f][k] * x_pad[t*HOP + k] |
// Pre-pass converts x and kernels to f16 in d_ws (fits L2 easily), so the
// steady-state K-loop is: 4x global b128 (A, L2-resident) + 4x ds_load_b128
// (B from LDS window) + 4x v_wmma_f32_16x16x32_f16.
// x window staged to LDS via global_load_async_to_lds_b128 when available.
// ---------------------------------------------------------------------------

typedef _Float16 half16 __attribute__((ext_vector_type(16)));
typedef _Float16 half8  __attribute__((ext_vector_type(8)));
typedef _Float16 half4  __attribute__((ext_vector_type(4)));
typedef float    float8 __attribute__((ext_vector_type(8)));
typedef float    f32x4  __attribute__((ext_vector_type(4)));
typedef int      v4i    __attribute__((ext_vector_type(4)));

#define HOP    480
#define NBINS  96
#define KTOT   7680            // 512+1024+2048+4096
#define KELEMS (KTOT * NBINS)  // 737280 floats per kernel bank
#define NT     2               // N-tiles of 16 frames per wave (A reuse x2)
#define FPB    (16 * NT)       // frames per block
#define NWAVES 6               // one wave per 16-bin M tile (6*16 = 96)
#define BLOCK  (NWAVES * 32)
#define MAX_WIN ((FPB - 1) * HOP + 4096)   // 18976 halves

#if defined(__has_builtin)
#  if __has_builtin(__builtin_amdgcn_global_load_async_to_lds_b128)
#    define HAVE_ASYNC_LDS 1
#  endif
#endif
#ifndef HAVE_ASYNC_LDS
#  define HAVE_ASYNC_LDS 0
#endif

// ---- pre-pass: f32 -> f16 conversion (runs once per launch; ~25 MB total) --
__global__ void TFST_cvt_f16(const float* __restrict__ src,
                             _Float16* __restrict__ dst, int n)
{
    const int i = (blockIdx.x * blockDim.x + threadIdx.x) * 4;
    if (i + 4 <= n) {
        const f32x4 v = *(const f32x4*)(src + i);
        half4 h;
#pragma unroll
        for (int e = 0; e < 4; ++e) h[e] = (_Float16)v[e];
        *(half4*)(dst + i) = h;
    } else {
        for (int j = i; j < n; ++j) dst[j] = (_Float16)src[j];
    }
}

// ---- main GEMM kernel ------------------------------------------------------
template <bool PRE>
__global__ __launch_bounds__(BLOCK)
void TFST_cqt_wmma(const float* __restrict__ x,
                   const _Float16* __restrict__ xh, int nx,
                   const float* __restrict__ kr,
                   const float* __restrict__ ki,
                   const _Float16* __restrict__ krh,
                   const _Float16* __restrict__ kih,
                   float* __restrict__ out, int T)
{
    __shared__ _Float16 s_x[MAX_WIN + 16];

    const int scale = blockIdx.y;
    const int size  = 512 << scale;          // 512,1024,2048,4096
    const int kcol  = size - 512;            // column offset within KTOT

    const int t0   = blockIdx.x * FPB;
    const int tid  = threadIdx.x;
    const int wave = tid >> 5;
    const int lane = tid & 31;

    // ---- Stage padded x window as f16 into LDS ----
    const int nwin = (FPB - 1) * HOP + size;
    const int base = t0 * HOP - (size >> 1);          // x index of s_x[0]
    if (PRE) {
#if HAVE_ASYNC_LDS
        if (base >= 0 && base + nwin <= nx) {
            // interior block: pure global->LDS copy, 16B chunks (nwin*2 % 16 == 0)
            typedef __attribute__((address_space(1))) v4i* gv4p;
            typedef __attribute__((address_space(3))) v4i* lv4p;
            gv4p gsrc = (gv4p)(xh + base);
            lv4p ldst = (lv4p)s_x;
            const int nchunks = (nwin * 2) >> 4;
            for (int c = tid; c < nchunks; c += BLOCK) {
                __builtin_amdgcn_global_load_async_to_lds_b128(
                    gsrc + c, ldst + c, 0, 0);
            }
            asm volatile("s_wait_asynccnt 0x0" ::: "memory");
        } else
#endif
        {
            for (int i = tid; i < nwin; i += BLOCK) {
                const int g = base + i;
                s_x[i] = (g >= 0 && g < nx) ? xh[g] : (_Float16)0.0f;
            }
        }
    } else {
        for (int i = tid; i < nwin; i += BLOCK) {
            const int g = base + i;
            const float v = (g >= 0 && g < nx) ? x[g] : 0.0f;
            s_x[i] = (_Float16)v;
        }
    }
    __syncthreads();

    // ---- Per-wave tile coordinates ----
    const int f0    = wave * 16;       // first bin of this wave's M tile
    const int m     = lane & 15;       // A row / B column within tile
    const int khalf = lane >> 4;       // 0: low K half-group, 1: high

    float8 acc_re[NT], acc_im[NT];
    const float8 zero = {0.f, 0.f, 0.f, 0.f, 0.f, 0.f, 0.f, 0.f};
#pragma unroll
    for (int j = 0; j < NT; ++j) { acc_re[j] = zero; acc_im[j] = zero; }

    const float*    krow_r  = kr  + (size_t)(f0 + m) * KTOT + kcol;
    const float*    krow_i  = ki  + (size_t)(f0 + m) * KTOT + kcol;
    const _Float16* krow_rh = krh + (size_t)(f0 + m) * KTOT + kcol;
    const _Float16* krow_ih = kih + (size_t)(f0 + m) * KTOT + kcol;

    // ---- K loop: 32 taps per WMMA ----
    for (int kb = 0; kb < size; kb += 32) {
        // contiguous A k order used here (matches B): lane<16 holds K=0..15,
        // lane>=16 holds K=16..31; element i -> VGPR i/2, half i%2.
        const int o0 = kb + khalf * 16;       // K = o0..o0+7
        const int o1 = kb + khalf * 16 + 8;   // K = o1..o1+7

        half16 Ar, Ai;
        if (PRE) {
            const half8 r0 = *(const half8*)(krow_rh + o0);
            const half8 r1 = *(const half8*)(krow_rh + o1);
            const half8 i0 = *(const half8*)(krow_ih + o0);
            const half8 i1 = *(const half8*)(krow_ih + o1);
#pragma unroll
            for (int e = 0; e < 8; ++e) {
                Ar[e] = r0[e]; Ar[8 + e] = r1[e];
                Ai[e] = i0[e]; Ai[8 + e] = i1[e];
            }
        } else {
            const f32x4 r0 = *(const f32x4*)(krow_r + o0);
            const f32x4 r1 = *(const f32x4*)(krow_r + o0 + 4);
            const f32x4 r2 = *(const f32x4*)(krow_r + o1);
            const f32x4 r3 = *(const f32x4*)(krow_r + o1 + 4);
            const f32x4 i0 = *(const f32x4*)(krow_i + o0);
            const f32x4 i1 = *(const f32x4*)(krow_i + o0 + 4);
            const f32x4 i2 = *(const f32x4*)(krow_i + o1);
            const f32x4 i3 = *(const f32x4*)(krow_i + o1 + 4);
#pragma unroll
            for (int e = 0; e < 4; ++e) {
                Ar[e]      = (_Float16)r0[e];
                Ar[4 + e]  = (_Float16)r1[e];
                Ar[8 + e]  = (_Float16)r2[e];
                Ar[12 + e] = (_Float16)r3[e];
                Ai[e]      = (_Float16)i0[e];
                Ai[4 + e]  = (_Float16)i1[e];
                Ai[8 + e]  = (_Float16)i2[e];
                Ai[12 + e] = (_Float16)i3[e];
            }
        }

#pragma unroll
        for (int j = 0; j < NT; ++j) {
            // B k order used here (contiguous, matches A): lane holds column n=lane&15,
            // K = khalf*16 + (0..15): 16 contiguous halves -> 2x ds_load_b128.
            const int n = lane & 15;
            const half8* bp =
                (const half8*)&s_x[(j * 16 + n) * HOP + kb + khalf * 16];
            const half8 blo = bp[0];
            const half8 bhi = bp[1];
            half16 B;
#pragma unroll
            for (int e = 0; e < 8; ++e) { B[e] = blo[e]; B[8 + e] = bhi[e]; }

            acc_re[j] = __builtin_amdgcn_wmma_f32_16x16x32_f16(
                false, Ar, false, B, (short)0, acc_re[j], false, false);
            acc_im[j] = __builtin_amdgcn_wmma_f32_16x16x32_f16(
                false, Ai, false, B, (short)0, acc_im[j], false, false);
        }
    }

    // ---- Epilogue: magnitude + store ----
    // C/D layout: lanes 0-15 -> N=lane,   M = v
    //             lanes16-31 -> N=lane-16, M = v+8
    const int frow = f0 + khalf * 8;
    const size_t out_base = (size_t)scale * NBINS * (size_t)T;
#pragma unroll
    for (int j = 0; j < NT; ++j) {
        const int t = t0 + j * 16 + (lane & 15);
        if (t < T) {
#pragma unroll
            for (int v = 0; v < 8; ++v) {
                const float re = acc_re[j][v];
                const float im = acc_im[j][v];
                out[out_base + (size_t)(frow + v) * T + t] =
                    __builtin_amdgcn_sqrtf(re * re + im * im);
            }
        }
    }
}

static inline size_t align256(size_t v) { return (v + 255) & ~(size_t)255; }

extern "C" void kernel_launch(void* const* d_in, const int* in_sizes, int n_in,
                              void* d_out, int out_size, void* d_ws, size_t ws_size,
                              hipStream_t stream) {
    (void)n_in;
    const float* x  = (const float*)d_in[0];
    const float* kr = (const float*)d_in[1];
    const float* ki = (const float*)d_in[2];
    float* out = (float*)d_out;

    const int nx = in_sizes[0];
    const int T  = out_size / (4 * NBINS);   // frames = 20001
    const int nbx = (T + FPB - 1) / FPB;
    dim3 grid(nbx, 4, 1);

    const size_t xh_bytes = align256((size_t)nx * 2);
    const size_t kh_bytes = align256((size_t)KELEMS * 2);
    const size_t need = xh_bytes + 2 * kh_bytes;   // ~22.3 MB

    if (ws_size >= need) {
        char* w = (char*)d_ws;
        _Float16* xh  = (_Float16*)w;
        _Float16* krh = (_Float16*)(w + xh_bytes);
        _Float16* kih = (_Float16*)(w + xh_bytes + kh_bytes);

        const int cvt_threads = 256;
        const int nbx_x = ((nx + 3) / 4 + cvt_threads - 1) / cvt_threads;
        const int nbx_k = ((KELEMS + 3) / 4 + cvt_threads - 1) / cvt_threads;
        TFST_cvt_f16<<<nbx_x, cvt_threads, 0, stream>>>(x, xh, nx);
        TFST_cvt_f16<<<nbx_k, cvt_threads, 0, stream>>>(kr, krh, KELEMS);
        TFST_cvt_f16<<<nbx_k, cvt_threads, 0, stream>>>(ki, kih, KELEMS);

        TFST_cqt_wmma<true><<<grid, BLOCK, 0, stream>>>(
            x, xh, nx, kr, ki, krh, kih, out, T);
    } else {
        TFST_cqt_wmma<false><<<grid, BLOCK, 0, stream>>>(
            x, nullptr, nx, kr, ki, nullptr, nullptr, out, T);
    }
}
